// SubstitutionLSTM_38208029065813
// MI455X (gfx1250) — matrix operand order repaired and checked by reading, not compiled, not run on hardware
//
#include <hip/hip_runtime.h>
#include <hip/hip_bf16.h>

// ---------------------------------------------------------------------------
// Model constants
// ---------------------------------------------------------------------------
#define VB   33
#define EE   256
#define HH   512
#define BB   64
#define SS   512
#define NBK  512
#define RIN  1120     // 2E + H + 96
#define MR   32768    // B*S rows

typedef __attribute__((ext_vector_type(16))) __bf16   v16bf;
typedef __attribute__((ext_vector_type(8)))  float    v8f;
typedef __attribute__((ext_vector_type(4)))  unsigned u32x4;

#define DEV __device__ __forceinline__

DEV unsigned short f2b(float f) {                       // f32 -> bf16 RNE
  unsigned u = __float_as_uint(f);
  u = u + 0x7fffu + ((u >> 16) & 1u);
  return (unsigned short)(u >> 16);
}

// A fragment: lane holds row m, elements k0+half*16+{0..15}
DEV v16bf ld_afrag(const unsigned short* rowk0, int half) {
  union { v16bf v; u32x4 q[2]; } u;
  u.q[0] = *reinterpret_cast<const u32x4*>(rowk0 + half * 16);
  u.q[1] = *reinterpret_cast<const u32x4*>(rowk0 + half * 16 + 8);
  return u.v;
}
// B fragment: lane holds 16 contiguous K elements of W row n
DEV v16bf ld_bfrag(const unsigned short* p) {
  union { v16bf v; u32x4 q[2]; } u;
  u.q[0] = *reinterpret_cast<const u32x4*>(p);
  u.q[1] = *reinterpret_cast<const u32x4*>(p + 8);
  return u.v;
}
DEV v8f wmma_bf16(v16bf a, v16bf b, v8f c) {
  return __builtin_amdgcn_wmma_f32_16x16x32_bf16(false, a, false, b, (short)0, c, false, false);
}
DEV float sigm(float x) { return 1.0f / (1.0f + expf(-x)); }

// device-wide generation barrier (persistent recurrence kernel)
DEV void gridbar(unsigned* cnt, unsigned* gen, unsigned nblk) {
  __syncthreads();
  if (threadIdx.x == 0) {
    __threadfence();
    volatile unsigned* vg = gen;
    unsigned g = *vg;
    if (atomicAdd(cnt, 1u) == nblk - 1u) {
      *(volatile unsigned*)cnt = 0u;
      __threadfence();
      atomicAdd(gen, 1u);
    } else {
      while (*vg == g) __builtin_amdgcn_s_sleep(1);
    }
  }
  __syncthreads();
  __threadfence();
}

// ---------------------------------------------------------------------------
// f32 -> bf16 convert
// ---------------------------------------------------------------------------
__global__ void cvt_bf16_kernel(const float* __restrict__ in,
                                unsigned short* __restrict__ out, int n) {
  int i = blockIdx.x * blockDim.x + threadIdx.x;
  if (i < n) out[i] = f2b(in[i]);
}

// ---------------------------------------------------------------------------
// Embedding gather: writes src_emb / tgt_emb columns of X1 (bf16)
// ---------------------------------------------------------------------------
__global__ void embed_kernel(const int* __restrict__ src, const int* __restrict__ tgt,
                             const unsigned short* __restrict__ embb,
                             unsigned short* __restrict__ X1) {
  int row = blockIdx.x;                 // b*S + s
  const unsigned short* se = embb + (size_t)src[row] * EE;
  const unsigned short* te = embb + (size_t)tgt[row] * EE;
  unsigned short* xr = X1 + (size_t)row * RIN;
  for (int i = threadIdx.x; i < EE; i += blockDim.x) {
    xr[i]      = se[i];
    xr[EE + i] = te[i];
  }
}

// ---------------------------------------------------------------------------
// n-gram histogram features + 2-layer MLPs; one block per batch row.
// ---------------------------------------------------------------------------
__global__ void ngram_kernel(const int* __restrict__ src,
    const float* __restrict__ uW1, const float* __restrict__ ub1,
    const float* __restrict__ uW2, const float* __restrict__ ub2,
    const float* __restrict__ bW1, const float* __restrict__ bb1,
    const float* __restrict__ bW2, const float* __restrict__ bb2,
    const float* __restrict__ tW1, const float* __restrict__ tb1,
    const float* __restrict__ tW2, const float* __restrict__ tb2,
    unsigned short* __restrict__ X1) {
  __shared__ int   ss[SS];
  __shared__ float hist[VB * VB];       // 1089, reused (first 512) for trigram
  __shared__ float uni[VB];
  __shared__ float hid[256];
  __shared__ float feat[96];
  __shared__ float cnts[3];

  int b = blockIdx.x, tid = threadIdx.x;
  for (int i = tid; i < SS; i += blockDim.x) ss[i] = src[b * SS + i];
  for (int i = tid; i < VB * VB; i += blockDim.x) hist[i] = 0.f;
  if (tid < VB) uni[tid] = 0.f;
  if (tid < 3)  cnts[tid] = 0.f;
  __syncthreads();

  // unigram
  for (int i = tid; i < SS; i += blockDim.x) {
    int v = ss[i];
    if (v > 0) { atomicAdd(&uni[v], 1.f); atomicAdd(&cnts[0], 1.f); }
  }
  __syncthreads();
  float lens = fmaxf(cnts[0], 1.f);
  if (tid < 64) {
    float s = ub1[tid];
    for (int v = 0; v < VB; ++v) s += (uni[v] / lens) * uW1[tid * VB + v];
    hid[tid] = fmaxf(s, 0.f);
  }
  __syncthreads();
  if (tid < 32) {
    float s = ub2[tid];
    for (int j = 0; j < 64; ++j) s += hid[j] * uW2[tid * 64 + j];
    feat[tid] = s;
  }
  __syncthreads();

  // bigram
  for (int i = tid; i < SS - 1; i += blockDim.x) {
    int c = ss[i], n2 = ss[i + 1];
    if (c > 0 && n2 > 0) { atomicAdd(&hist[c * VB + n2], 1.f); atomicAdd(&cnts[1], 1.f); }
  }
  __syncthreads();
  {
    float bs = 1.f / fmaxf(cnts[1], 1.f);
    float s = 0.f;
    for (int i = 0; i < VB * VB; ++i) s += hist[i] * bW1[(size_t)tid * (VB * VB) + i];
    hid[tid] = fmaxf(s * bs + bb1[tid], 0.f);
  }
  __syncthreads();
  if (tid < 32) {
    float s = bb2[tid];
    for (int j = 0; j < 256; ++j) s += hid[j] * bW2[tid * 256 + j];
    feat[32 + tid] = s;
  }
  __syncthreads();

  // trigram (reuse hist[0..511])
  for (int i = tid; i < NBK; i += blockDim.x) hist[i] = 0.f;
  __syncthreads();
  for (int i = tid; i < SS - 2; i += blockDim.x) {
    int c1 = ss[i], c2 = ss[i + 1], c3 = ss[i + 2];
    if (c1 > 0 && c2 > 0 && c3 > 0) {
      int idx = (c1 * (VB * VB) + c2 * VB + c3) & (NBK - 1);
      atomicAdd(&hist[idx], 1.f); atomicAdd(&cnts[2], 1.f);
    }
  }
  __syncthreads();
  {
    float ts = 1.f / fmaxf(cnts[2], 1.f);
    float s = 0.f;
    for (int i = 0; i < NBK; ++i) s += hist[i] * tW1[(size_t)tid * NBK + i];
    hid[tid] = fmaxf(s * ts + tb1[tid], 0.f);
  }
  __syncthreads();
  if (tid < 32) {
    float s = tb2[tid];
    for (int j = 0; j < 256; ++j) s += hid[j] * tW2[tid * 256 + j];
    feat[64 + tid] = s;
  }
  __syncthreads();

  // broadcast gstats to every time step of this batch row
  for (int i = tid; i < SS * 96; i += blockDim.x) {
    int s0 = i / 96, j = i % 96;
    X1[((size_t)(b * SS + s0)) * RIN + (RIN - 96) + j] = f2b(feat[j]);
  }
}

// ---------------------------------------------------------------------------
// bf16 WMMA GEMM, 2x2 tile blocking per wave (4 independent accumulators):
//   Out[M x N](f32) = A[M x K](bf16,lda) @ W[NW x K]^T + b0 + b1
// Lanes whose W row is out of range (n >= NW) read a zeroed dummy row, so the
// inner loop has no divergence; stores are guarded at NW.
// ---------------------------------------------------------------------------
__global__ __launch_bounds__(256) void gemm_bf16_kernel(
    const unsigned short* __restrict__ A, int lda, int K,
    const unsigned short* __restrict__ W, int NW,
    const unsigned short* __restrict__ zrow,
    const float* __restrict__ b0, const float* __restrict__ b1,
    float* __restrict__ Out, int ldo, int M, int N) {
  const int lane = threadIdx.x & 31;
  const int wv   = threadIdx.x >> 5;
  const int MT2 = M >> 5, NT2 = N >> 5;       // 32x32 quads
  long quad = (long)blockIdx.x * (blockDim.x >> 5) + wv;
  if (quad >= (long)MT2 * NT2) return;
  const int ntq = (int)(quad % NT2), mtq = (int)(quad / NT2);
  const int col = lane & 15, half = lane >> 4;
  const int n0 = ntq * 32 + col, n1 = n0 + 16;

  float bias0 = 0.f, bias1 = 0.f;
  if (n0 < NW) { if (b0) bias0 += b0[n0]; if (b1) bias0 += b1[n0]; }
  if (n1 < NW) { if (b0) bias1 += b0[n1]; if (b1) bias1 += b1[n1]; }
  v8f c00, c01, c10, c11;
  #pragma unroll
  for (int r = 0; r < 8; ++r) { c00[r] = bias0; c01[r] = bias1; c10[r] = bias0; c11[r] = bias1; }

  const unsigned short* arow0 = A + (size_t)(mtq * 32 + col) * lda;
  const unsigned short* arow1 = arow0 + (size_t)16 * lda;
  const unsigned short* wr0 = (n0 < NW) ? (W + (size_t)n0 * K) : zrow;
  const unsigned short* wr1 = (n1 < NW) ? (W + (size_t)n1 * K) : zrow;

  for (int k0 = 0; k0 < K; k0 += 32) {
    v16bf a0 = ld_afrag(arow0 + k0, half);
    v16bf a1 = ld_afrag(arow1 + k0, half);
    v16bf bf0 = ld_bfrag(wr0 + k0 + half * 16);
    v16bf bf1 = ld_bfrag(wr1 + k0 + half * 16);
    c00 = wmma_bf16(a0, bf0, c00);
    c01 = wmma_bf16(a0, bf1, c01);
    c10 = wmma_bf16(a1, bf0, c10);
    c11 = wmma_bf16(a1, bf1, c11);
  }

  const int mrow0 = mtq * 32 + (lane < 16 ? 0 : 8);
  const int mrow1 = mrow0 + 16;
  if (n0 < NW) {
    #pragma unroll
    for (int r = 0; r < 8; ++r) {
      Out[(size_t)(mrow0 + r) * ldo + n0] = c00[r];
      Out[(size_t)(mrow1 + r) * ldo + n0] = c10[r];
    }
  }
  if (n1 < NW) {
    #pragma unroll
    for (int r = 0; r < 8; ++r) {
      Out[(size_t)(mrow0 + r) * ldo + n1] = c01[r];
      Out[(size_t)(mrow1 + r) * ldo + n1] = c11[r];
    }
  }
}

// ---------------------------------------------------------------------------
// Persistent LSTM recurrence. XG holds x@Wih^T + bih + bhh for all (b,s).
// Each wave owns one (mt, nt) output tile and all 4 gate fragments, so the
// gate nonlinearity is lane-local; c stays in VGPRs for the whole scan.
// Whh rows needed by this block (4 gates x 2 n-tiles) live in LDS.
// One grid barrier per time step.
// ---------------------------------------------------------------------------
__global__ __launch_bounds__(256) void lstm_recur_kernel(
    const float* __restrict__ XG, int hid,
    const unsigned short* __restrict__ Whh,
    unsigned short* __restrict__ Hstate,
    unsigned short* __restrict__ Outp, int outLd, int outColOff,
    int reverse, unsigned* __restrict__ barCnt,
    unsigned* __restrict__ barGen, int nblk) {
  extern __shared__ unsigned short sW[];   // [4 gates][32 cols][hid]
  const int lane = threadIdx.x & 31;
  const int wid  = blockIdx.x * (blockDim.x >> 5) + (threadIdx.x >> 5);
  const int col = lane & 15, half = lane >> 4, moff = half * 8;
  const int mt = wid & 3, nt = wid >> 2;
  const int ntBase = blockIdx.x * 2;
  const int g4 = hid * 4;

  // preload this block's Whh slice into LDS (one time)
  for (int i = threadIdx.x; i < 128 * hid; i += blockDim.x) {
    int lr = i / hid, k = i - lr * hid;
    int g = lr >> 5, jj = lr & 31;
    sW[i] = Whh[((size_t)(g * hid + ntBase * 16 + jj)) * hid + k];
  }
  // zero h state
  for (int i = blockIdx.x * blockDim.x + threadIdx.x; i < BB * hid; i += nblk * blockDim.x)
    Hstate[i] = 0;
  gridbar(barCnt, barGen, (unsigned)nblk);

  const int n  = nt * 16 + col;
  const int ln = (nt - ntBase) * 16 + col;   // local column in LDS slice
  const unsigned short* swi = sW + ((size_t)(0 * 32 + ln)) * hid;
  const unsigned short* swf = sW + ((size_t)(1 * 32 + ln)) * hid;
  const unsigned short* swg = sW + ((size_t)(2 * 32 + ln)) * hid;
  const unsigned short* swo = sW + ((size_t)(3 * 32 + ln)) * hid;
  const unsigned short* arow = Hstate + (size_t)(mt * 16 + col) * hid;

  float creg[8];
  #pragma unroll
  for (int r = 0; r < 8; ++r) creg[r] = 0.f;

  for (int t = 0; t < SS; ++t) {
    const int s = reverse ? (SS - 1 - t) : t;
    v8f ci, cf, cg, co;
    #pragma unroll
    for (int r = 0; r < 8; ++r) {
      size_t base = ((size_t)(mt * 16 + moff + r) * SS + s) * g4;
      ci[r] = XG[base + 0 * hid + n];
      cf[r] = XG[base + 1 * hid + n];
      cg[r] = XG[base + 2 * hid + n];
      co[r] = XG[base + 3 * hid + n];
    }
    for (int k0 = 0; k0 < hid; k0 += 32) {
      v16bf a   = ld_afrag(arow + k0, half);
      v16bf bi  = ld_bfrag(swi + k0 + half * 16);
      v16bf bf_ = ld_bfrag(swf + k0 + half * 16);
      v16bf bg  = ld_bfrag(swg + k0 + half * 16);
      v16bf bo  = ld_bfrag(swo + k0 + half * 16);
      ci = wmma_bf16(a, bi, ci);
      cf = wmma_bf16(a, bf_, cf);
      cg = wmma_bf16(a, bg, cg);
      co = wmma_bf16(a, bo, co);
    }
    #pragma unroll
    for (int r = 0; r < 8; ++r) {
      float iv = sigm(ci[r]), fv = sigm(cf[r]);
      float gv = tanhf(cg[r]), ov = sigm(co[r]);
      float cc = fv * creg[r] + iv * gv;
      creg[r] = cc;
      unsigned short hb = f2b(ov * tanhf(cc));
      int m = mt * 16 + moff + r;
      Hstate[(size_t)m * hid + n] = hb;
      Outp[((size_t)m * SS + s) * outLd + outColOff + n] = hb;
    }
    gridbar(barCnt, barGen, (unsigned)nblk);
  }
}

// ---------------------------------------------------------------------------
// Host side
// ---------------------------------------------------------------------------
// input order (setup_inputs insertion order, params depth-first):
//  0 src, 1 tgt_input, 2 embedding,
//  3 enc_f.Wih, 4 enc_f.Whh, 5 enc_f.bih, 6 enc_f.bhh,
//  7 enc_b.Wih, 8 enc_b.Whh, 9 enc_b.bih, 10 enc_b.bhh,
// 11 uni_W1, 12 uni_b1, 13 uni_W2, 14 uni_b2,
// 15 big_W1, 16 big_b1, 17 big_W2, 18 big_b2,
// 19 tri_W1, 20 tri_b1, 21 tri_W2, 22 tri_b2,
// 23..26 rnn0 (Wih,Whh,bih,bhh), 27..30 rnn1, 31..34 rnn2,
// 35 fc_W, 36 fc_b

// bf16 weight pool offsets (elements)
static const size_t EF_WIH = 0;
static const size_t EF_WHH = EF_WIH + (size_t)1024 * 256;
static const size_t EB_WIH = EF_WHH + (size_t)1024 * 256;
static const size_t EB_WHH = EB_WIH + (size_t)1024 * 256;
static const size_t R0_WIH = EB_WHH + (size_t)1024 * 256;
static const size_t R0_WHH = R0_WIH + (size_t)2048 * 1120;
static const size_t R1_WIH = R0_WHH + (size_t)2048 * 512;
static const size_t R1_WHH = R1_WIH + (size_t)2048 * 512;
static const size_t R2_WIH = R1_WHH + (size_t)2048 * 512;
static const size_t R2_WHH = R2_WIH + (size_t)2048 * 512;
static const size_t FC_WO  = R2_WHH + (size_t)2048 * 512;
static const size_t EMB_O  = FC_WO + (size_t)33 * 512;
static const size_t WBF_ELEMS = EMB_O + (size_t)33 * 256;

extern "C" void kernel_launch(void* const* d_in, const int* in_sizes, int n_in,
                              void* d_out, int out_size, void* d_ws, size_t ws_size,
                              hipStream_t stream) {
  (void)in_sizes; (void)n_in; (void)out_size; (void)ws_size;
  auto al = [](size_t x) { return (x + 255) & ~(size_t)255; };

  // [0,256): barrier counters; [256,4096): zero dummy W row (K_max=1120 bf16)
  const size_t OFF_BAR = 0;
  const size_t OFF_ZRO = 256;
  const size_t OFF_WBF = 4096;
  const size_t OFF_X1  = al(OFF_WBF + WBF_ELEMS * 2);
  const size_t OFF_XG  = al(OFF_X1 + (size_t)MR * RIN * 2);
  const size_t OFF_HA  = al(OFF_XG + (size_t)MR * 2048 * 4);
  const size_t OFF_HB  = OFF_HA + (size_t)MR * 512 * 2;
  const size_t OFF_HS  = OFF_HB + (size_t)MR * 512 * 2;

  char* ws = (char*)d_ws;
  unsigned*       bar = (unsigned*)(ws + OFF_BAR);
  unsigned short* zro = (unsigned short*)(ws + OFF_ZRO);
  unsigned short* wbf = (unsigned short*)(ws + OFF_WBF);
  unsigned short* X1  = (unsigned short*)(ws + OFF_X1);
  float*          XG  = (float*)(ws + OFF_XG);
  float*          XGb = XG + (size_t)MR * 1024;
  unsigned short* HA  = (unsigned short*)(ws + OFF_HA);
  unsigned short* HB  = (unsigned short*)(ws + OFF_HB);
  unsigned short* HS  = (unsigned short*)(ws + OFF_HS);

  const int* src = (const int*)d_in[0];
  const int* tgt = (const int*)d_in[1];
  auto F = [&](int i) { return (const float*)d_in[i]; };

  (void)hipFuncSetAttribute((const void*)lstm_recur_kernel,
                            hipFuncAttributeMaxDynamicSharedMemorySize, 131072);

  hipMemsetAsync(ws, 0, 4096, stream);   // barrier counters + zero row

  auto cvt = [&](int idx, size_t off, int n) {
    cvt_bf16_kernel<<<(n + 255) / 256, 256, 0, stream>>>(F(idx), wbf + off, n);
  };
  cvt(3,  EF_WIH, 1024 * 256);  cvt(4,  EF_WHH, 1024 * 256);
  cvt(7,  EB_WIH, 1024 * 256);  cvt(8,  EB_WHH, 1024 * 256);
  cvt(23, R0_WIH, 2048 * 1120); cvt(24, R0_WHH, 2048 * 512);
  cvt(27, R1_WIH, 2048 * 512);  cvt(28, R1_WHH, 2048 * 512);
  cvt(31, R2_WIH, 2048 * 512);  cvt(32, R2_WHH, 2048 * 512);
  cvt(35, FC_WO,  33 * 512);    cvt(2,  EMB_O,  33 * 256);

  embed_kernel<<<MR, 256, 0, stream>>>(src, tgt, wbf + EMB_O, X1);
  ngram_kernel<<<BB, 256, 0, stream>>>(src,
      F(11), F(12), F(13), F(14), F(15), F(16), F(17), F(18),
      F(19), F(20), F(21), F(22), X1);

  auto gemm = [&](const unsigned short* A, int lda, int K, const unsigned short* W,
                  int NW, const float* b0, const float* b1, float* Out, int ldo, int N) {
    long quads = (long)(MR / 32) * (N / 32);
    int blocks = (int)((quads + 7) / 8);
    gemm_bf16_kernel<<<blocks, 256, 0, stream>>>(A, lda, K, W, NW, zro, b0, b1,
                                                 Out, ldo, MR, N);
  };
  auto recur = [&](const float* xg, int hid, const unsigned short* whh,
                   unsigned short* outp, int outLd, int outColOff, int rev, int nblk) {
    size_t lds = (size_t)4 * 32 * hid * 2;
    lstm_recur_kernel<<<nblk, 256, lds, stream>>>(xg, hid, whh, HS, outp, outLd,
                                                  outColOff, rev, bar, bar + 1, nblk);
  };

  // bidirectional encoder (Hh=256, gates=1024); ctx written into X1 cols 512..1023
  gemm(X1, RIN, 256, wbf + EF_WIH, 1024, F(5), F(6), XG,  1024, 1024);
  gemm(X1, RIN, 256, wbf + EB_WIH, 1024, F(9), F(10), XGb, 1024, 1024);
  recur(XG,  256, wbf + EF_WHH, X1, RIN, 512, 0, 8);
  recur(XGb, 256, wbf + EB_WHH, X1, RIN, 768, 1, 8);

  // decoder stack (H=512, gates=2048)
  gemm(X1, RIN, RIN, wbf + R0_WIH, 2048, F(25), F(26), XG, 2048, 2048);
  recur(XG, 512, wbf + R0_WHH, HA, 512, 0, 0, 16);
  gemm(HA, 512, 512, wbf + R1_WIH, 2048, F(29), F(30), XG, 2048, 2048);
  recur(XG, 512, wbf + R1_WHH, HB, 512, 0, 0, 16);
  gemm(HB, 512, 512, wbf + R2_WIH, 2048, F(33), F(34), XG, 2048, 2048);
  recur(XG, 512, wbf + R2_WHH, HA, 512, 0, 0, 16);

  // final FC 512 -> 33 (N padded to 64, guarded; OOB lanes read the zero row)
  gemm(HA, 512, 512, wbf + FC_WO, 33, F(36), nullptr, (float*)d_out, 33, 64);
}
